// CrossAttnMem_q_aug_85220741087739
// MI455X (gfx1250) — matrix operand order repaired and checked by reading, not compiled, not run on hardware
//
#include <hip/hip_runtime.h>
#include <hip/hip_bf16.h>

typedef __bf16 bf16;
typedef bf16  v16bf __attribute__((ext_vector_type(16)));
typedef bf16  v8bf  __attribute__((ext_vector_type(8)));
typedef float v8f   __attribute__((ext_vector_type(8)));
typedef int   v4i   __attribute__((ext_vector_type(4)));

typedef __attribute__((address_space(1))) v4i gv4i;  // global int4
typedef __attribute__((address_space(3))) v4i lv4i;  // LDS int4

#if __has_builtin(__builtin_amdgcn_global_load_async_to_lds_b128) && \
    __has_builtin(__builtin_amdgcn_s_wait_asynccnt)
#define HAVE_ASYNC_LDS 1
#else
#define HAVE_ASYNC_LDS 0
#endif

#define WMMA_BF16(a, b, c) \
  __builtin_amdgcn_wmma_f32_16x16x32_bf16(false, (a), false, (b), (short)0, (c), false, false)

static __device__ __forceinline__ v8f vzero8() {
  v8f z;
#pragma unroll
  for (int i = 0; i < 8; ++i) z[i] = 0.f;
  return z;
}

static __device__ __forceinline__ v16bf join8(v8bf lo, v8bf hi) {
  v16bf r;
#pragma unroll
  for (int i = 0; i < 8; ++i) { r[i] = lo[i]; r[8 + i] = hi[i]; }
  return r;
}

static __device__ __forceinline__ v8bf f2bf8(v8f x) {
  v8bf r;
#pragma unroll
  for (int i = 0; i < 8; ++i) r[i] = (bf16)x[i];
  return r;
}

// A fragment (16x32, M = lane&15) from an f32 row pointer: halves 0..7 = K[half*8..+7],
// halves 8..15 = K[16+half*8..+7]   (per CDNA5 ISA 16-bit A layout)
static __device__ __forceinline__ v16bf frag_a_f32(const float* row, int k, int half) {
  v8f lo = *(const v8f*)(row + k + half * 8);
  v8f hi = *(const v8f*)(row + k + 16 + half * 8);
  return join8(f2bf8(lo), f2bf8(hi));
}

// B fragment (32x16, N = lane&15) from an f32 row (NT layout: row indexed by N, contiguous K):
// halves 0..7 = K[half*8..+7], halves 8..15 = K[16+half*8..+7]
static __device__ __forceinline__ v16bf frag_b_f32(const float* row, int k, int half) {
  v8f lo = *(const v8f*)(row + k + half * 8);
  v8f hi = *(const v8f*)(row + k + 16 + half * 8);
  return join8(f2bf8(lo), f2bf8(hi));
}

static __device__ __forceinline__ v16bf frag_b_bf16(const bf16* row, int k, int half) {
  v8bf lo = *(const v8bf*)(row + k + half * 8);
  v8bf hi = *(const v8bf*)(row + k + 16 + half * 8);
  return join8(lo, hi);
}

// ---------------------------------------------------------------------------
// Phase 1a: q = emb_l @ Wq^T -> qT[b][c][n] ; k = emb_u @ Wk^T -> kT[d][n]
// Orientation: A = emb (M = n), B = W (N = c). K = 64 (2 WMMA steps).
// ---------------------------------------------------------------------------
__global__ __launch_bounds__(256) void proj_qk_kernel(const float* __restrict__ emb,
                                                      const float* __restrict__ Wq,
                                                      const float* __restrict__ Wk,
                                                      bf16* __restrict__ qT,
                                                      bf16* __restrict__ kT) {
  const int z = blockIdx.z;  // 0..7 -> q batches, 8..15 -> k batches
  const float* src;
  const float* W;
  bf16* dst;
  if (z < 8) {
    src = emb + (size_t)(8 + z) * 1024 * 64;  // emb_l
    W = Wq;
    dst = qT + (size_t)z * 512 * 1024;
  } else {
    const int zz = z - 8;
    src = emb + (size_t)zz * 1024 * 64;       // emb_u
    W = Wk;
    dst = kT + (size_t)zz * 512 * 1024;
  }
  const int w = blockIdx.x * 8 + (threadIdx.x >> 5);  // 0..2047 = 64 n-tiles x 32 c-tiles
  const int lane = threadIdx.x & 31;
  const int l15 = lane & 15, half = lane >> 4;
  const int nt = w >> 5, ct = w & 31;
  const int n0 = nt * 16, c0 = ct * 16;

  const float* arow = src + (size_t)(n0 + l15) * 64;
  const float* brow = W + (size_t)(c0 + l15) * 64;

  v8f acc = vzero8();
#pragma unroll
  for (int k = 0; k < 64; k += 32) {
    v16bf a = frag_a_f32(arow, k, half);
    v16bf b = frag_b_f32(brow, k, half);
    acc = WMMA_BF16(a, b, acc);
  }
  // C: N=lane&15 -> c, M=i(+8*half) -> n : per lane 8 contiguous n values
  bf16* drow = dst + (size_t)(c0 + l15) * 1024 + n0 + half * 8;
  *(v8bf*)drow = f2bf8(acc);
}

// ---------------------------------------------------------------------------
// Phase 1b: v = emb_u @ Wv^T -> vbank[n][d], d = b*512 + c
// Orientation: A = Wv (M = c), B = emb (N = n) so C lanes write contiguous d.
// ---------------------------------------------------------------------------
__global__ __launch_bounds__(256) void proj_v_kernel(const float* __restrict__ emb,
                                                     const float* __restrict__ Wv,
                                                     bf16* __restrict__ vbank) {
  const int b = blockIdx.z;  // 0..7 (emb_u)
  const float* src = emb + (size_t)b * 1024 * 64;
  const int w = blockIdx.x * 8 + (threadIdx.x >> 5);  // 32 c-tiles x 64 n-tiles
  const int lane = threadIdx.x & 31;
  const int l15 = lane & 15, half = lane >> 4;
  const int ct = w >> 6, nt = w & 63;
  const int c0 = ct * 16, n0 = nt * 16;

  const float* arow = Wv + (size_t)(c0 + l15) * 64;
  const float* brow = src + (size_t)(n0 + l15) * 64;

  v8f acc = vzero8();
#pragma unroll
  for (int k = 0; k < 64; k += 32) {
    v16bf a = frag_a_f32(arow, k, half);
    v16bf b = frag_b_f32(brow, k, half);
    acc = WMMA_BF16(a, b, acc);
  }
  // C: N -> n, M -> c : per lane 8 contiguous c (=d) values
  bf16* drow = vbank + (size_t)(n0 + l15) * 4096 + b * 512 + c0 + half * 8;
  *(v8bf*)drow = f2bf8(acc);
}

// ---------------------------------------------------------------------------
// NT GEMM: C[b][m][n'] = sum_k A[b][m][k] * B[n'][k], bf16 in, bf16 out (f32 acc).
//   STATS : accumulate per-batch sum/sumsq of C into stats[b*2 +{0,1}]
//   TRANSC: store C transposed, C[b][n'][m] (contiguous in m -> 16B stores)
// Block tile 128x128x32, 8 waves (2x4 grid of 16x16 WMMA tiles per wave).
// Double-buffered LDS; tiles staged with GLOBAL_LOAD_ASYNC_TO_LDS_B128
// (ASYNCcnt-tracked, no VGPR round-trip), fallback to sync loads.
// ---------------------------------------------------------------------------
template <int M, int N, int K, bool STATS, bool TRANSC>
__global__ __launch_bounds__(256) void gemm_nt_kernel(const bf16* __restrict__ A,
                                                      const bf16* __restrict__ B,
                                                      bf16* __restrict__ C,
                                                      float* __restrict__ stats) {
  constexpr int BM = 128, BN = 128, BK = 32, PITCH = 40;
  __shared__ bf16 As[2][BM][PITCH];
  __shared__ bf16 Bs[2][BN][PITCH];
  __shared__ float red[256];

  const int b = blockIdx.z;
  const int m0 = blockIdx.y * BM;
  const int n0 = blockIdx.x * BN;
  A += (size_t)b * M * K;
  C += (size_t)b * M * N;

  const int t = threadIdx.x;
  const int wid = t >> 5, lane = t & 31;
  const int wm = wid & 3, wn = wid >> 2;  // 4 (M) x 2 (N) waves -> 32x64 per wave
  const int l15 = lane & 15, half = lane >> 4;

  v8f acc[2][4];
#pragma unroll
  for (int i = 0; i < 2; ++i)
#pragma unroll
    for (int j = 0; j < 4; ++j) acc[i][j] = vzero8();

  // Each thread stages two 16B chunks of A and two of B per tile:
  // rows (t>>2) and (t>>2)+64, chunk (t&3)*8. 4 async instructions per wave.
  const int sr = t >> 2;
  const int sc8 = (t & 3) * 8;
  auto stage = [&](int buf, int k0) {
#pragma unroll
    for (int i = 0; i < 2; ++i) {
      const int r = sr + i * 64;
      bf16* ga = const_cast<bf16*>(A + (size_t)(m0 + r) * K + k0 + sc8);
      bf16* gb = const_cast<bf16*>(B + (size_t)(n0 + r) * K + k0 + sc8);
#if HAVE_ASYNC_LDS
      __builtin_amdgcn_global_load_async_to_lds_b128((gv4i*)ga, (lv4i*)&As[buf][r][sc8],
                                                     0, 0);
      __builtin_amdgcn_global_load_async_to_lds_b128((gv4i*)gb, (lv4i*)&Bs[buf][r][sc8],
                                                     0, 0);
#else
      *(v8bf*)&As[buf][r][sc8] = *(const v8bf*)ga;
      *(v8bf*)&Bs[buf][r][sc8] = *(const v8bf*)gb;
#endif
    }
  };

  constexpr int NSTEP = K / BK;
  stage(0, 0);
  for (int s = 0; s < NSTEP; ++s) {
    const int cur = s & 1;
    if (s + 1 < NSTEP) {
      stage(cur ^ 1, (s + 1) * BK);  // background prefetch into other buffer
#if HAVE_ASYNC_LDS
      __builtin_amdgcn_s_wait_asynccnt(4);  // oldest 4 (current buffer) done
#endif
    } else {
#if HAVE_ASYNC_LDS
      __builtin_amdgcn_s_wait_asynccnt(0);  // drain
#endif
    }
    __syncthreads();

    v16bf af[2], bfr[4];
#pragma unroll
    for (int i = 0; i < 2; ++i) {
      const int r = wm * 32 + i * 16 + l15;
      af[i] = join8(*(const v8bf*)&As[cur][r][half * 8],
                    *(const v8bf*)&As[cur][r][16 + half * 8]);
    }
#pragma unroll
    for (int j = 0; j < 4; ++j) {
      const int r = wn * 64 + j * 16 + l15;
      bfr[j] = join8(*(const v8bf*)&Bs[cur][r][half * 8],
                     *(const v8bf*)&Bs[cur][r][16 + half * 8]);
    }
#pragma unroll
    for (int i = 0; i < 2; ++i)
#pragma unroll
      for (int j = 0; j < 4; ++j) acc[i][j] = WMMA_BF16(af[i], bfr[j], acc[i][j]);
    __syncthreads();  // protect buffer 'cur' from next prefetch overwrite
  }

  if constexpr (STATS) {
    float s = 0.f, s2 = 0.f;
#pragma unroll
    for (int i = 0; i < 2; ++i)
#pragma unroll
      for (int j = 0; j < 4; ++j)
#pragma unroll
        for (int v = 0; v < 8; ++v) {
          const float x = acc[i][j][v];
          s += x;
          s2 += x * x;
        }
    red[t] = s;
    __syncthreads();
    for (int o = 128; o > 0; o >>= 1) {
      if (t < o) red[t] += red[t + o];
      __syncthreads();
    }
    if (t == 0) atomicAdd(&stats[b * 2 + 0], red[0]);
    __syncthreads();
    red[t] = s2;
    __syncthreads();
    for (int o = 128; o > 0; o >>= 1) {
      if (t < o) red[t] += red[t + o];
      __syncthreads();
    }
    if (t == 0) atomicAdd(&stats[b * 2 + 1], red[0]);
  }

#pragma unroll
  for (int i = 0; i < 2; ++i) {
    const int mrow = m0 + wm * 32 + i * 16 + half * 8;  // + v
#pragma unroll
    for (int j = 0; j < 4; ++j) {
      const int ncol = n0 + wn * 64 + j * 16 + l15;
      if constexpr (TRANSC) {
        *(v8bf*)(C + (size_t)ncol * M + mrow) = f2bf8(acc[i][j]);
      } else {
#pragma unroll
        for (int v = 0; v < 8; ++v) C[(size_t)(mrow + v) * N + ncol] = (bf16)acc[i][j][v];
      }
    }
  }
}

// ---------------------------------------------------------------------------
// Fused layernorm-softmax, one block per (b,c) row of 4096, in-place bf16.
// mean cancels under softmax: attn = softmax(inv_std * (s - rowmax)).
// ---------------------------------------------------------------------------
__global__ __launch_bounds__(256) void softmax_kernel(bf16* __restrict__ S,
                                                      const float* __restrict__ stats) {
  const int r = blockIdx.x;  // 0..4095
  const int b = r >> 9;
  bf16* row = S + (size_t)r * 4096;
  const int t = threadIdx.x;
  __shared__ float red[256];

  float v[16];
  {
    v8bf x0 = *(const v8bf*)(row + t * 16);
    v8bf x1 = *(const v8bf*)(row + t * 16 + 8);
#pragma unroll
    for (int i = 0; i < 8; ++i) { v[i] = (float)x0[i]; v[8 + i] = (float)x1[i]; }
  }
  float m = -3.4e38f;
#pragma unroll
  for (int i = 0; i < 16; ++i) m = fmaxf(m, v[i]);
  red[t] = m;
  __syncthreads();
  for (int o = 128; o > 0; o >>= 1) {
    if (t < o) red[t] = fmaxf(red[t], red[t + o]);
    __syncthreads();
  }
  m = red[0];
  __syncthreads();

  const float CNT = 512.f * 4096.f;
  const float mean = stats[b * 2 + 0] / CNT;
  const float var = stats[b * 2 + 1] / CNT - mean * mean;
  const float inv = rsqrtf(var + 1e-5f);

  float s = 0.f;
#pragma unroll
  for (int i = 0; i < 16; ++i) {
    v[i] = __expf((v[i] - m) * inv);
    s += v[i];
  }
  red[t] = s;
  __syncthreads();
  for (int o = 128; o > 0; o >>= 1) {
    if (t < o) red[t] += red[t + o];
    __syncthreads();
  }
  const float scale = 1.f / red[0];

  v8bf y0, y1;
#pragma unroll
  for (int i = 0; i < 8; ++i) {
    y0[i] = (bf16)(v[i] * scale);
    y1[i] = (bf16)(v[8 + i] * scale);
  }
  *(v8bf*)(row + t * 16) = y0;
  *(v8bf*)(row + t * 16 + 8) = y1;
}

// ---------------------------------------------------------------------------
// Phase 5: out[b][n][e] = sum_c Wo[e][c] * ctxb[b][n][c].
// A = Wo (M = e, f32->bf16 on the fly), B = ctxb row (N = n, contiguous c).
// ---------------------------------------------------------------------------
__global__ __launch_bounds__(256) void out_proj_kernel(const bf16* __restrict__ ctxb,
                                                       const float* __restrict__ Wo,
                                                       float* __restrict__ out) {
  const int w = blockIdx.x * 8 + (threadIdx.x >> 5);  // 0..2047 = 8b x 4e x 64n
  const int lane = threadIdx.x & 31;
  const int l15 = lane & 15, half = lane >> 4;
  const int b = w >> 8;
  const int rem = w & 255;
  const int e0 = (rem >> 6) * 16;
  const int n0 = (rem & 63) * 16;

  const float* arow = Wo + (size_t)(e0 + l15) * 512;
  const bf16* brow = ctxb + ((size_t)b * 1024 + n0 + l15) * 512;

  v8f acc = vzero8();
#pragma unroll
  for (int k = 0; k < 512; k += 32) {
    v16bf a = frag_a_f32(arow, k, half);
    v16bf bb = frag_b_bf16(brow, k, half);
    acc = WMMA_BF16(a, bb, acc);
  }
  // C: N -> n, M -> e : per lane 8 contiguous f32 (one 32B store)
  float* dst = out + ((size_t)b * 1024 + n0 + l15) * 64 + e0 + half * 8;
  *(v8f*)dst = acc;
}

__global__ void init_stats_kernel(float* __restrict__ s) {
  if (threadIdx.x < 16) s[threadIdx.x] = 0.f;
}

// ---------------------------------------------------------------------------
extern "C" void kernel_launch(void* const* d_in, const int* in_sizes, int n_in,
                              void* d_out, int out_size, void* d_ws, size_t ws_size,
                              hipStream_t stream) {
  (void)in_sizes; (void)n_in; (void)out_size; (void)ws_size;
  const float* emb = (const float*)d_in[0];
  const float* Wq = (const float*)d_in[1];
  const float* Wk = (const float*)d_in[2];
  const float* Wv = (const float*)d_in[3];
  const float* Wo = (const float*)d_in[4];
  float* out = (float*)d_out;

  char* ws = (char*)d_ws;
  size_t off = 0;
  auto carve = [&](size_t bytes) -> char* {
    char* p = ws + off;
    off += (bytes + 255) & ~(size_t)255;
    return p;
  };
  bf16* qT     = (bf16*)carve(8ull * 512 * 1024 * 2);   // (b, c, n)
  bf16* kT     = (bf16*)carve(4096ull * 1024 * 2);      // (d, n)
  bf16* vbank  = (bf16*)carve(1024ull * 4096 * 2);      // (n, d)
  bf16* scores = (bf16*)carve(8ull * 512 * 4096 * 2);   // (b, c, d) -> attn in place
  bf16* ctxb   = (bf16*)carve(8ull * 1024 * 512 * 2);   // (b, n, c)
  float* stats = (float*)carve(64);

  init_stats_kernel<<<1, 32, 0, stream>>>(stats);
  proj_qk_kernel<<<dim3(256, 1, 16), 256, 0, stream>>>(emb, Wq, Wk, qT, kT);
  proj_v_kernel<<<dim3(256, 1, 8), 256, 0, stream>>>(emb, Wv, vbank);
  // scores[b,c,d] = qT[b] (512x1024) x kT (4096x1024)^T ; stats on the fly
  gemm_nt_kernel<512, 4096, 1024, true, false>
      <<<dim3(32, 4, 8), 256, 0, stream>>>(qT, kT, scores, stats);
  softmax_kernel<<<4096, 256, 0, stream>>>(scores, stats);
  // ctx[b,n,c] = attn[b] (512x4096) x vbank (1024x4096)^T, stored transposed
  gemm_nt_kernel<512, 1024, 4096, false, true>
      <<<dim3(8, 4, 8), 256, 0, stream>>>(scores, vbank, ctxb, nullptr);
  out_proj_kernel<<<256, 256, 0, stream>>>(ctxb, Wo, out);
}
